// GATNet_88613765251553
// MI455X (gfx1250) — matrix operand order repaired and checked by reading, not compiled, not run on hardware
//
#include <hip/hip_runtime.h>
#include <math.h>

// ---------------------------------------------------------------------------
// GAT layer for MI455X (gfx1250, wave32, WMMA).
//   h = x @ W           -> bf16 WMMA GEMM, f32 accumulate (software-pipelined)
//   a_src/a_dst = h@att -> wave-per-row reduction
//   edge softmax        -> atomic max / atomic add segment ops
//   scatter h[src]*alpha-> global_atomic_add_f32
//   out = elu(out+bias)
// ---------------------------------------------------------------------------

#define IN_C 512
#define HID  128
#define NEG_SLOPE 0.2f

typedef __attribute__((ext_vector_type(16))) __bf16 v16bf;
typedef __attribute__((ext_vector_type(8)))  float  v8f;

union Frag {
    v16bf v;
    unsigned int u[8];
};

__device__ __forceinline__ unsigned int pack_bf16(float lo, float hi) {
    // round-to-nearest-even fp32 -> bf16, packed {hi,lo}
    unsigned int a = __float_as_uint(lo);
    unsigned int b = __float_as_uint(hi);
    a += 0x7FFFu + ((a >> 16) & 1u);
    b += 0x7FFFu + ((b >> 16) & 1u);
    return (a >> 16) | (b & 0xFFFF0000u);
}

__device__ __forceinline__ void atomicMaxF(float* addr, float v) {
    if (v >= 0.0f) atomicMax((int*)addr, __float_as_int(v));
    else           atomicMin((unsigned int*)addr, __float_as_uint(v));
}

// ---------------------------------------------------------------------------
// 0) init: zero d_out, set segment-max to -inf, zero denominators
// ---------------------------------------------------------------------------
__global__ void init_kernel(float* __restrict__ out, float* __restrict__ maxb,
                            float* __restrict__ denom, int N) {
    int t = blockIdx.x * blockDim.x + threadIdx.x;
    long total = (long)N * HID;
    if (t < total) out[t] = 0.0f;
    if (t < N) {
        maxb[t]  = __uint_as_float(0xFF800000u); // -inf
        denom[t] = 0.0f;
    }
}

// ---------------------------------------------------------------------------
// 1) pack W (512x128 f32, row-major) into bf16 B-fragment layout.
// Fragment layout (V_WMMA_F32_16X16X32_BF16, B = 32x16):
//   dword index dw = ((kb*8 + cb)*32 + lane)*8 + d
//   hi = lane>>4, n = cb*16 + (lane&15), K0 = kb*32 + (d>>2)*16 + hi*8 + 2*(d&3)
//   dword = { bf16(W[K0+1][n]) , bf16(W[K0][n]) }
// ---------------------------------------------------------------------------
__global__ void wpack_kernel(const float* __restrict__ W, unsigned int* __restrict__ wfrag) {
    int dw = blockIdx.x * blockDim.x + threadIdx.x;
    if (dw >= 16 * 8 * 32 * 8) return;
    int kb   = dw >> 11;
    int rem  = dw & 2047;
    int cb   = rem >> 8;
    int lane = (rem >> 3) & 31;
    int d    = rem & 7;
    int hi   = lane >> 4;
    int n    = cb * 16 + (lane & 15);
    int K0   = kb * 32 + (d >> 2) * 16 + hi * 8 + 2 * (d & 3);
    float f0 = W[(size_t)K0 * HID + n];
    float f1 = W[(size_t)(K0 + 1) * HID + n];
    wfrag[dw] = pack_bf16(f0, f1);
}

// ---------------------------------------------------------------------------
// 2) GEMM: h[N x 128] = x[N x 512] @ W. 256 threads = 8 waves; block computes
// a 128x128 tile; wave computes 16 rows x 128 cols via 8 f32 accumulators.
// Double-buffered LDS staging (global loads overlapped with WMMAs) plus a
// two-deep rotating register buffer for B fragments, so the wait before each
// WMMA is "dscnt <= 2" (next pair still in flight) instead of a full LDS
// round-trip to zero.
// ---------------------------------------------------------------------------
__global__ __launch_bounds__(256) void gemm_kernel(const float* __restrict__ x,
                                                   const unsigned int* __restrict__ wfrag,
                                                   float* __restrict__ h, int N) {
    __shared__ unsigned int ldsW[2][2048]; // double-buffered kb-slab (2 x 8 KB)

    const int tid  = threadIdx.x;
    const int wave = tid >> 5;
    const int lane = tid & 31;
    const int hi   = lane >> 4;
    const int lo   = lane & 15;
    const int row0 = blockIdx.x * 128 + wave * 16;

    int rowA = row0 + lo;
    if (rowA >= N) rowA = N - 1;               // clamp reads; stores are guarded
    const float* xrow = x + (size_t)rowA * IN_C;

    v8f acc[8];
#pragma unroll
    for (int cb = 0; cb < 8; ++cb) acc[cb] = (v8f){0.f,0.f,0.f,0.f,0.f,0.f,0.f,0.f};

    // ---- prologue: stage slab 0 into LDS buffer 0, preload A tile for kb=0
    {
        const uint4* s4 = (const uint4*)(wfrag + tid * 8);
        uint4* d4 = (uint4*)(&ldsW[0][tid * 8]);
        d4[0] = s4[0];
        d4[1] = s4[1];
    }
    float4 cf0, cf1, cf2, cf3;
    {
        const float4* g0 = (const float4*)(xrow + hi * 8);
        const float4* g1 = (const float4*)(xrow + 16 + hi * 8);
        cf0 = g0[0]; cf1 = g0[1]; cf2 = g1[0]; cf3 = g1[1];
    }
    __syncthreads();

    for (int kb = 0; kb < 16; ++kb) {
        const int cur = kb & 1;
        const bool has_next = (kb < 15);

        // ---- issue next-slab global loads early (hidden behind the WMMAs)
        uint4 st0, st1;
        float4 nf0, nf1, nf2, nf3;
        if (has_next) {
            const uint4* s4 = (const uint4*)(wfrag + (size_t)(kb + 1) * 2048 + tid * 8);
            st0 = s4[0];
            st1 = s4[1];
            const float4* g0 = (const float4*)(xrow + (kb + 1) * 32 + hi * 8);
            const float4* g1 = (const float4*)(xrow + (kb + 1) * 32 + 16 + hi * 8);
            nf0 = g0[0]; nf1 = g0[1]; nf2 = g1[0]; nf3 = g1[1];
        }
        if (kb < 14) __builtin_prefetch(xrow + (kb + 2) * 32 + hi * 8, 0, 1);

        // ---- build A fragment (16x32 bf16) from current-tile registers
        Frag a;
        a.u[0] = pack_bf16(cf0.x, cf0.y);
        a.u[1] = pack_bf16(cf0.z, cf0.w);
        a.u[2] = pack_bf16(cf1.x, cf1.y);
        a.u[3] = pack_bf16(cf1.z, cf1.w);
        a.u[4] = pack_bf16(cf2.x, cf2.y);
        a.u[5] = pack_bf16(cf2.z, cf2.w);
        a.u[6] = pack_bf16(cf3.x, cf3.y);
        a.u[7] = pack_bf16(cf3.z, cf3.w);

        // ---- B fragments: two-deep rotating register pipeline over cb.
        // Loads for cb+1 are issued BEFORE the WMMA consuming cb, so the
        // hardware wait before each WMMA leaves one ds_load pair in flight.
        const unsigned int* wl = &ldsW[cur][lane * 8];
        uint4 pb0[2], pb1[2];
        {
            const uint4* ls = (const uint4*)(wl);
            pb0[0] = ls[0];
            pb1[0] = ls[1];
        }
#pragma unroll
        for (int cb = 0; cb < 8; ++cb) {
            const int pc = cb & 1;
            const int pn = pc ^ 1;
            if (cb < 7) {
                const uint4* ls = (const uint4*)(wl + (cb + 1) * 256);
                pb0[pn] = ls[0];
                pb1[pn] = ls[1];
            }
            Frag b;
            b.u[0] = pb0[pc].x; b.u[1] = pb0[pc].y; b.u[2] = pb0[pc].z; b.u[3] = pb0[pc].w;
            b.u[4] = pb1[pc].x; b.u[5] = pb1[pc].y; b.u[6] = pb1[pc].z; b.u[7] = pb1[pc].w;
            acc[cb] = __builtin_amdgcn_wmma_f32_16x16x32_bf16(
                false, a.v, false, b.v, (short)0, acc[cb], false, false);
        }

        // ---- commit next slab to the other LDS buffer, rotate A registers
        if (has_next) {
            uint4* d4 = (uint4*)(&ldsW[cur ^ 1][tid * 8]);
            d4[0] = st0;
            d4[1] = st1;
            cf0 = nf0; cf1 = nf1; cf2 = nf2; cf3 = nf3;
        }
        __syncthreads();
    }

    // C/D layout: VGPR r, lanes 0-15 -> M=r, N=lane; lanes 16-31 -> M=r+8, N=lane-16
#pragma unroll
    for (int cb = 0; cb < 8; ++cb) {
        int col = cb * 16 + lo;
#pragma unroll
        for (int r = 0; r < 8; ++r) {
            int row = row0 + r + hi * 8;
            if (row < N) h[(size_t)row * HID + col] = acc[cb][r];
        }
    }
}

// ---------------------------------------------------------------------------
// 3) per-row dots: a_src = h . att_src, a_dst = h . att_dst (wave per row)
// ---------------------------------------------------------------------------
__global__ void rowdot_kernel(const float* __restrict__ h,
                              const float* __restrict__ att_s, const float* __restrict__ att_d,
                              float* __restrict__ a_s, float* __restrict__ a_d, int N) {
    int g = blockIdx.x * blockDim.x + threadIdx.x;
    int row = g >> 5;
    int lane = g & 31;
    if (row >= N) return;
    float4 hv = ((const float4*)(h + (size_t)row * HID))[lane];
    float4 sv = ((const float4*)att_s)[lane];
    float4 dv = ((const float4*)att_d)[lane];
    float ps = hv.x * sv.x + hv.y * sv.y + hv.z * sv.z + hv.w * sv.w;
    float pd = hv.x * dv.x + hv.y * dv.y + hv.z * dv.z + hv.w * dv.w;
#pragma unroll
    for (int off = 16; off > 0; off >>= 1) {
        ps += __shfl_xor(ps, off, 32);
        pd += __shfl_xor(pd, off, 32);
    }
    if (lane == 0) { a_s[row] = ps; a_d[row] = pd; }
}

// ---------------------------------------------------------------------------
// 4) edge logits + segment max (edges E..E+N-1 are self loops)
// ---------------------------------------------------------------------------
__global__ void logits_kernel(const int* __restrict__ ei,
                              const float* __restrict__ a_s, const float* __restrict__ a_d,
                              float* __restrict__ eval, float* __restrict__ maxb,
                              int E, int EN) {
    int t = blockIdx.x * blockDim.x + threadIdx.x;
    if (t >= EN) return;
    int s, d;
    if (t < E) { s = ei[t]; d = ei[E + t]; } else { s = d = t - E; }
    float e = a_s[s] + a_d[d];
    e = e > 0.0f ? e : NEG_SLOPE * e;
    eval[t] = e;
    atomicMaxF(maxb + d, e);
}

// ---------------------------------------------------------------------------
// 5) exp(e - max) + segment sum (in-place over eval)
// ---------------------------------------------------------------------------
__global__ void expsum_kernel(const int* __restrict__ ei, float* __restrict__ eval,
                              const float* __restrict__ maxb, float* __restrict__ denom,
                              int E, int EN) {
    int t = blockIdx.x * blockDim.x + threadIdx.x;
    if (t >= EN) return;
    int d = (t < E) ? ei[E + t] : (t - E);
    float ex = expf(eval[t] - maxb[d]);
    eval[t] = ex;
    atomicAdd(denom + d, ex);
}

// ---------------------------------------------------------------------------
// 6) weighted scatter: out[d] += alpha * h[s]  (wave per edge, float4 gather)
// ---------------------------------------------------------------------------
__global__ void scatter_kernel(const int* __restrict__ ei, const float* __restrict__ eexp,
                               const float* __restrict__ denom, const float* __restrict__ h,
                               float* __restrict__ out, int E, int EN) {
    int g = blockIdx.x * blockDim.x + threadIdx.x;
    int edge = g >> 5;
    int lane = g & 31;
    if (edge >= EN) return;
    int s, d;
    if (edge < E) { s = ei[edge]; d = ei[E + edge]; } else { s = d = edge - E; }
    float alpha = eexp[edge] / (denom[d] + 1e-16f);
    float4 hv = ((const float4*)(h + (size_t)s * HID))[lane];
    float* ob = out + (size_t)d * HID + lane * 4;
    atomicAdd(ob + 0, hv.x * alpha);
    atomicAdd(ob + 1, hv.y * alpha);
    atomicAdd(ob + 2, hv.z * alpha);
    atomicAdd(ob + 3, hv.w * alpha);
}

// ---------------------------------------------------------------------------
// 7) finalize: out = elu(out + bias)
// ---------------------------------------------------------------------------
__global__ void finalize_kernel(float* __restrict__ out, const float* __restrict__ bias, int N) {
    long t = (long)blockIdx.x * blockDim.x + threadIdx.x;
    if (t >= (long)N * HID) return;
    float v = out[t] + bias[t & (HID - 1)];
    out[t] = v > 0.0f ? v : expf(v) - 1.0f;
}

// ---------------------------------------------------------------------------
extern "C" void kernel_launch(void* const* d_in, const int* in_sizes, int n_in,
                              void* d_out, int out_size, void* d_ws, size_t ws_size,
                              hipStream_t stream) {
    (void)n_in; (void)out_size; (void)ws_size;
    const float* x       = (const float*)d_in[0];
    const int*   ei      = (const int*)d_in[1];
    const float* W       = (const float*)d_in[2];
    const float* att_src = (const float*)d_in[3];
    const float* att_dst = (const float*)d_in[4];
    const float* bias    = (const float*)d_in[5];
    float* out = (float*)d_out;

    const int N  = in_sizes[0] / IN_C;
    const int E  = in_sizes[1] / 2;
    const int EN = E + N;

    // workspace layout (floats)
    float* ws     = (float*)d_ws;
    float* h      = ws;                               // N*128
    float* a_s    = h + (size_t)N * HID;              // N
    float* a_d    = a_s + N;                          // N
    float* maxb   = a_d + N;                          // N
    float* denom  = maxb + N;                         // N
    float* eval   = denom + N;                        // E+N
    unsigned int* wfrag = (unsigned int*)(eval + EN); // 32768 dwords (128 KB)

    const int T = 256;
    long nh = (long)N * HID;

    init_kernel<<<(int)((nh + T - 1) / T), T, 0, stream>>>(out, maxb, denom, N);
    wpack_kernel<<<(16 * 8 * 32 * 8) / T, T, 0, stream>>>(W, wfrag);
    gemm_kernel<<<(N + 127) / 128, T, 0, stream>>>(x, wfrag, h, N);
    rowdot_kernel<<<(int)(((long)N * 32 + T - 1) / T), T, 0, stream>>>(h, att_src, att_dst, a_s, a_d, N);
    logits_kernel<<<(EN + T - 1) / T, T, 0, stream>>>(ei, a_s, a_d, eval, maxb, E, EN);
    expsum_kernel<<<(EN + T - 1) / T, T, 0, stream>>>(ei, eval, maxb, denom, E, EN);
    scatter_kernel<<<(int)(((long)EN * 32 + T - 1) / T), T, 0, stream>>>(ei, eval, denom, h, out, E, EN);
    finalize_kernel<<<(int)((nh + T - 1) / T), T, 0, stream>>>(out, bias, N);
}
